// ResidualAttentionBlock_41635412967910
// MI455X (gfx1250) — matrix operand order repaired and checked by reading, not compiled, not run on hardware
//
#include <hip/hip_runtime.h>

// ---------------------------------------------------------------------------
// MI455X (gfx1250) transformer block: LN1 -> QKV -> flash-attn -> out_proj(+res)
//                                     -> LN2 -> FC+quickGELU -> proj(+res)
// bf16 WMMA (v_wmma_f32_16x16x32_bf16) with fp32 accumulate for all GEMMs;
// fused flash attention (no [B,H,S,S] tensor); TDM tensor_load_to_lds for
// K/V tile staging in attention.
// ---------------------------------------------------------------------------

typedef __bf16 bf16;
typedef __attribute__((ext_vector_type(16))) __bf16 v16bf;
typedef __attribute__((ext_vector_type(8)))  __bf16 v8bf;
typedef __attribute__((ext_vector_type(8)))  float  v8f;
typedef __attribute__((ext_vector_type(4)))  unsigned int v4u;
typedef __attribute__((ext_vector_type(8)))  int v8i;
typedef __attribute__((ext_vector_type(4)))  int v4i;

#define DMODEL 768
#define NHEAD  12
#define DHEAD  64
#define DFF    3072
#define SEQ    1024
#define BATCH  8
#define NTOK   (SEQ * BATCH)

// ---------------------------------------------------------------------------
// WMMA wrapper: D = A(16x32 bf16) x B(32x16 bf16) + C(16x16 f32)
__device__ __forceinline__ v8f wmma_bf16(v16bf a, v16bf b, v8f c) {
  return __builtin_amdgcn_wmma_f32_16x16x32_bf16(
      /*neg_a=*/false, a, /*neg_b=*/false, b,
      /*c_mod=*/(short)0, c, /*reuse_a=*/false, /*reuse_b=*/false);
}

// A-fragment loader (contiguous K order):
//   lane L holds row m = L&15; halves 0..15 cover K = 2*kgrp..2*kgrp+15,
//   kgrp = 8*(L>=16).
__device__ __forceinline__ v16bf load_afrag(const bf16* rowPtr, int kgrp) {
  v8bf lo = *(const v8bf*)(rowPtr + 2 * kgrp);
  v8bf hi = *(const v8bf*)(rowPtr + 2 * kgrp + 8);
  v16bf r;
#pragma unroll
  for (int i = 0; i < 8; ++i) { r[i] = lo[i]; r[i + 8] = hi[i]; }
  return r;
}

// ---------------------------------------------------------------------------
// TDM 2D tile load: global (row-major, 2-byte elems, strided rows) -> LDS
// (packed). Builds D# groups per CDNA5 ISA ch.8; one instruction per wave,
// tracked by TENSORcnt.
__device__ __forceinline__ void tdm_load_2d(unsigned lds_off, const void* gptr,
                                            unsigned width_elems,
                                            unsigned height_rows,
                                            unsigned long long row_stride_elems) {
  unsigned long long ga = (unsigned long long)(size_t)gptr;
  v4u g0;
  g0[0] = 1u;                                          // count=1 (valid user D#)
  g0[1] = lds_off;                                     // lds_addr (bytes)
  g0[2] = (unsigned)(ga & 0xffffffffu);                // global_addr[31:0]
  g0[3] = (unsigned)((ga >> 32) & 0x01ffffffu)         // global_addr[56:32]
          | (2u << 30);                                // type = 2 ("image")
  v8i g1;
  g1[0] = (int)(1u << 16);                             // data_size=1 (2 bytes)
  g1[1] = (int)((width_elems & 0xffffu) << 16);        // tensor_dim0[15:0]
  g1[2] = (int)((width_elems >> 16) & 0xffffu)         // tensor_dim0[31:16]
          | (int)((height_rows & 0xffffu) << 16);      // tensor_dim1[15:0]
  g1[3] = (int)((height_rows >> 16) & 0xffffu)         // tensor_dim1[31:16]
          | (int)((width_elems & 0xffffu) << 16);      // tile_dim0
  g1[4] = (int)(height_rows & 0xffffu);                // tile_dim1 (tile_dim2=0)
  g1[5] = (int)(row_stride_elems & 0xffffffffu);       // dim0_stride[31:0]
  g1[6] = (int)((row_stride_elems >> 32) & 0xffffu);   // dim0_stride[47:32]
  g1[7] = 0;                                           // dim1_stride (unused 2D)
  v4i gz = {0, 0, 0, 0};
#if __has_include(<hip/amd_detail/amd_gfx1250_TDM.h>)
  v8i gz8 = {0, 0, 0, 0, 0, 0, 0, 0};
  __builtin_amdgcn_tensor_load_to_lds(g0, g1, gz, gz, gz8, 0);  // clang-23 form
#else
  __builtin_amdgcn_tensor_load_to_lds(g0, g1, gz, gz, 0);       // ROCm 7.2 form
#endif
}

// ---------------------------------------------------------------------------
// fp32 -> bf16 elementwise (weight conversion, once per launch)
__global__ void cvt_f32_bf16(const float* __restrict__ in,
                             bf16* __restrict__ out, int n) {
  int i = blockIdx.x * blockDim.x + threadIdx.x;
  if (i < n) out[i] = (bf16)in[i];
}

// ---------------------------------------------------------------------------
// LayerNorm over D=768, one wave32 per row, bf16 output.
__global__ __launch_bounds__(256)
void ln_bf16(const float* __restrict__ x, const float* __restrict__ g,
             const float* __restrict__ bta, bf16* __restrict__ out, int nrows) {
  int row = blockIdx.x * 8 + (threadIdx.x >> 5);
  int lane = threadIdx.x & 31;
  if (row >= nrows) return;
  const float* xr = x + (size_t)row * DMODEL;
  float v[24];
  float s = 0.f, s2 = 0.f;
#pragma unroll
  for (int j = 0; j < 24; ++j) {
    v[j] = xr[lane + j * 32];
    s += v[j];
    s2 += v[j] * v[j];
  }
#pragma unroll
  for (int m = 16; m >= 1; m >>= 1) {
    s += __shfl_xor(s, m, 32);
    s2 += __shfl_xor(s2, m, 32);
  }
  float mu = s * (1.0f / DMODEL);
  float var = s2 * (1.0f / DMODEL) - mu * mu;
  float rstd = rsqrtf(var + 1e-5f);
  bf16* orow = out + (size_t)row * DMODEL;
#pragma unroll
  for (int j = 0; j < 24; ++j) {
    int c = lane + j * 32;
    orow[c] = (bf16)((v[j] - mu) * rstd * g[c] + bta[c]);
  }
}

// ---------------------------------------------------------------------------
// GEMM: C[M,N] = A[M,K](bf16) * W[N,K]^T(bf16) + bias, fused epilogue.
// MODE 0: f32 out | 1: bf16 out | 2: bf16 quickGELU out | 3: f32 out + residual
// Block = 256 thr = 8 waves (2x4); wave tile 32x64 (2x4 WMMA); block tile 64x256.
template <int MODE>
__global__ __launch_bounds__(256)
void gemm_bf16(const bf16* __restrict__ A, const bf16* __restrict__ W,
               const float* __restrict__ bias, const float* __restrict__ res,
               void* __restrict__ outv, int M, int N, int K) {
  const int tid = threadIdx.x;
  const int lane = tid & 31;
  const int wid = tid >> 5;
  const int wm = wid >> 2;   // 0..1
  const int wn = wid & 3;    // 0..3
  const int hw = lane >> 4;  // half-wave
  const int l16 = lane & 15;

  const int rowBase = blockIdx.y * 64 + wm * 32;
  const int colBase = blockIdx.x * 256 + wn * 64;

  const v8f zero = {};
  v8f acc[2][4];
#pragma unroll
  for (int a = 0; a < 2; ++a)
#pragma unroll
    for (int b = 0; b < 4; ++b) acc[a][b] = zero;

  const int kgrp = hw * 8;   // A-frag K sub-offset
  const int kb = hw * 16;    // B-frag K sub-offset

  const bf16* Arow[2];
  Arow[0] = A + (size_t)(rowBase + l16) * K;
  Arow[1] = A + (size_t)(rowBase + 16 + l16) * K;
  const bf16* Wrow[4];
#pragma unroll
  for (int t = 0; t < 4; ++t) Wrow[t] = W + (size_t)(colBase + t * 16 + l16) * K;

  for (int k0 = 0; k0 < K; k0 += 32) {
    __builtin_prefetch(Wrow[0] + k0 + 256, 0, 1);
    v16bf af[2];
#pragma unroll
    for (int t = 0; t < 2; ++t) af[t] = load_afrag(Arow[t] + k0, kgrp);
    v16bf bf[4];
#pragma unroll
    for (int t = 0; t < 4; ++t) bf[t] = *(const v16bf*)(Wrow[t] + k0 + kb);
#pragma unroll
    for (int tm = 0; tm < 2; ++tm)
#pragma unroll
      for (int tn = 0; tn < 4; ++tn)
        acc[tm][tn] = wmma_bf16(af[tm], bf[tn], acc[tm][tn]);
  }

  float* outF = (float*)outv;
  bf16* outB = (bf16*)outv;
#pragma unroll
  for (int tm = 0; tm < 2; ++tm) {
#pragma unroll
    for (int tn = 0; tn < 4; ++tn) {
      int col = colBase + tn * 16 + l16;
      float bv = bias[col];
#pragma unroll
      for (int r = 0; r < 8; ++r) {
        // C/D layout: lane half 0 -> rows 0..7, half 1 -> rows 8..15
        int row = rowBase + tm * 16 + hw * 8 + r;
        size_t idx = (size_t)row * N + col;
        float v = acc[tm][tn][r] + bv;
        if (MODE == 0) {
          outF[idx] = v;
        } else if (MODE == 1) {
          outB[idx] = (bf16)v;
        } else if (MODE == 2) {
          float gl = v / (1.0f + __expf(-1.702f * v));  // quickGELU
          outB[idx] = (bf16)gl;
        } else {
          outF[idx] = v + res[idx];
        }
      }
    }
  }
}

// ---------------------------------------------------------------------------
// Flash attention. qkv:[NTOK, 3*D] bf16 (token = s*B + b), ctx:[NTOK, D] bf16.
// Block = 128 thr = 4 waves; block owns (b, head, 64 queries); each wave owns
// 16 queries. K/V tiles (32 keys) staged into LDS by the Tensor Data Mover;
// online softmax; P x V accumulated with WMMA.
__global__ __launch_bounds__(128)
void flash_attn(const bf16* __restrict__ qkv, bf16* __restrict__ ctx) {
  const int qblk = blockIdx.x;  // 64-query block
  const int hd = blockIdx.y;    // head
  const int b = blockIdx.z;     // batch
  const int tid = threadIdx.x;
  const int lane = tid & 31;
  const int w = tid >> 5;
  const int hw = lane >> 4;
  const int l16 = lane & 15;

  __shared__ __align__(32) unsigned short Kt[32][64];     // [key][feat] (TDM)
  __shared__ __align__(32) unsigned short Vs[32][64];     // [key][dh]   (TDM)
  __shared__ __align__(32) unsigned short Vt[64][32];     // [dh][key]
  __shared__ __align__(32) unsigned short Pt[4][16][32];  // per-wave P tile

  // Persistent Q A-fragments (16 queries x 64 features = 2 frags)
  const int qs = qblk * 64 + w * 16 + l16;
  const bf16* qrow = qkv + (size_t)(qs * BATCH + b) * (3 * DMODEL) + hd * DHEAD;
  const int kgrp = hw * 8;
  v16bf aq0 = load_afrag(qrow, kgrp);
  v16bf aq1 = load_afrag(qrow + 32, kgrp);

  const v8f zero = {};
  v8f acc[4];
  float mrow[8], lrow[8];
#pragma unroll
  for (int g = 0; g < 4; ++g) acc[g] = zero;
#pragma unroll
  for (int r = 0; r < 8; ++r) { mrow[r] = -1e30f; lrow[r] = 0.f; }

  const float LOG2E = 1.44269504088896f;
  const float SCALE = 0.125f;  // 1/sqrt(64)

  for (int kt0 = 0; kt0 < SEQ; kt0 += 32) {
    // --- TDM: DMA K and V tiles (32 keys x 64 feats, row stride 8*2304) ---
    if (w == 0) {
      size_t tok0 = (size_t)(kt0 * BATCH + b) * (3 * DMODEL);
      tdm_load_2d((unsigned)(size_t)&Kt[0][0],
                  qkv + tok0 + DMODEL + hd * DHEAD, 64, 32,
                  (unsigned long long)BATCH * 3 * DMODEL);
      tdm_load_2d((unsigned)(size_t)&Vs[0][0],
                  qkv + tok0 + 2 * DMODEL + hd * DHEAD, 64, 32,
                  (unsigned long long)BATCH * 3 * DMODEL);
    }
    __builtin_amdgcn_s_wait_tensorcnt(0);
    __syncthreads();

    // --- transpose V tile in LDS: Vs[key][dh] -> Vt[dh][key] ---
#pragma unroll
    for (int i = 0; i < 8; ++i) {
      int lin = i * 128 + tid;  // dword index 0..1023
      int key = lin >> 5;
      int fd = lin & 31;
      unsigned int vv = *(const unsigned int*)&Vs[key][fd * 2];
      Vt[fd * 2][key] = (unsigned short)(vv & 0xffffu);
      Vt[fd * 2 + 1][key] = (unsigned short)(vv >> 16);
    }
    __syncthreads();

    // --- scores: two 16-key sub-tiles ---
    float st[2][8];
#pragma unroll
    for (int t = 0; t < 2; ++t) {
      v16bf bk0 = *(const v16bf*)&Kt[t * 16 + l16][hw * 16];
      v16bf bk1 = *(const v16bf*)&Kt[t * 16 + l16][32 + hw * 16];
      v8f sc = zero;
      sc = wmma_bf16(aq0, bk0, sc);
      sc = wmma_bf16(aq1, bk1, sc);
#pragma unroll
      for (int r = 0; r < 8; ++r) st[t][r] = sc[r] * SCALE;
    }

    // --- online softmax update (row = hw*8 + r lives in one 16-lane half) ---
#pragma unroll
    for (int r = 0; r < 8; ++r) {
      float mt = fmaxf(st[0][r], st[1][r]);
#pragma unroll
      for (int m = 8; m >= 1; m >>= 1) mt = fmaxf(mt, __shfl_xor(mt, m, 32));
      float mnew = fmaxf(mrow[r], mt);
      float alpha = exp2f((mrow[r] - mnew) * LOG2E);
      float p0 = exp2f((st[0][r] - mnew) * LOG2E);
      float p1 = exp2f((st[1][r] - mnew) * LOG2E);
      Pt[w][hw * 8 + r][l16] = __builtin_bit_cast(unsigned short, (bf16)p0);
      Pt[w][hw * 8 + r][16 + l16] = __builtin_bit_cast(unsigned short, (bf16)p1);
      float rs = p0 + p1;
#pragma unroll
      for (int m = 8; m >= 1; m >>= 1) rs += __shfl_xor(rs, m, 32);
      lrow[r] = lrow[r] * alpha + rs;
      mrow[r] = mnew;
#pragma unroll
      for (int g = 0; g < 4; ++g) acc[g][r] *= alpha;
    }
    __syncthreads();

    // --- ctx += P(16x32) x V(32x64): A-frag from Pt, B-frags from Vt ---
    v16bf ap = load_afrag((const bf16*)&Pt[w][l16][0], kgrp);
#pragma unroll
    for (int g = 0; g < 4; ++g) {
      v16bf bv = *(const v16bf*)&Vt[g * 16 + l16][hw * 16];
      acc[g] = wmma_bf16(ap, bv, acc[g]);
    }
    __syncthreads();  // before next tile's TDM overwrites Kt/Vs
  }

  // --- epilogue: normalize and scatter to ctx[token, hd*64 + col] ---
#pragma unroll
  for (int g = 0; g < 4; ++g) {
#pragma unroll
    for (int r = 0; r < 8; ++r) {
      int s = qblk * 64 + w * 16 + hw * 8 + r;
      size_t tok = (size_t)(s * BATCH + b);
      float v = acc[g][r] / lrow[r];
      ctx[tok * DMODEL + hd * DHEAD + g * 16 + l16] = (bf16)v;
    }
  }
}

// ---------------------------------------------------------------------------
extern "C" void kernel_launch(void* const* d_in, const int* in_sizes, int n_in,
                              void* d_out, int out_size, void* d_ws,
                              size_t ws_size, hipStream_t stream) {
  (void)in_sizes; (void)n_in; (void)out_size; (void)ws_size;

  const float* x     = (const float*)d_in[0];
  const float* in_w  = (const float*)d_in[1];
  const float* in_b  = (const float*)d_in[2];
  const float* out_w = (const float*)d_in[3];
  const float* out_b = (const float*)d_in[4];
  const float* ln1g  = (const float*)d_in[5];
  const float* ln1b  = (const float*)d_in[6];
  const float* fc_w  = (const float*)d_in[7];
  const float* fc_b  = (const float*)d_in[8];
  const float* pr_w  = (const float*)d_in[9];
  const float* pr_b  = (const float*)d_in[10];
  const float* ln2g  = (const float*)d_in[11];
  const float* ln2b  = (const float*)d_in[12];

  char* ws = (char*)d_ws;
  size_t off = 0;
  auto alloc = [&](size_t bytes) {
    void* p = ws + off;
    off += (bytes + 255) & ~(size_t)255;
    return p;
  };
  bf16* Wqkv = (bf16*)alloc((size_t)3 * DMODEL * DMODEL * 2);
  bf16* Wout = (bf16*)alloc((size_t)DMODEL * DMODEL * 2);
  bf16* Wfc  = (bf16*)alloc((size_t)DFF * DMODEL * 2);
  bf16* Wpr  = (bf16*)alloc((size_t)DMODEL * DFF * 2);
  bf16* hbuf = (bf16*)alloc((size_t)NTOK * DMODEL * 2);      // LN1 / LN2 output
  bf16* qkvB = (bf16*)alloc((size_t)NTOK * 3 * DMODEL * 2);  // qkv
  bf16* ctxB = (bf16*)alloc((size_t)NTOK * DMODEL * 2);      // attention output
  float* x1  = (float*)alloc((size_t)NTOK * DMODEL * 4);     // post-attn residual
  bf16* fcA  = (bf16*)alloc((size_t)NTOK * DFF * 2);         // gelu(fc) acts

  // 1) weights fp32 -> bf16
  {
    int n;
    n = 3 * DMODEL * DMODEL;
    cvt_f32_bf16<<<(n + 255) / 256, 256, 0, stream>>>(in_w, Wqkv, n);
    n = DMODEL * DMODEL;
    cvt_f32_bf16<<<(n + 255) / 256, 256, 0, stream>>>(out_w, Wout, n);
    n = DFF * DMODEL;
    cvt_f32_bf16<<<(n + 255) / 256, 256, 0, stream>>>(fc_w, Wfc, n);
    n = DMODEL * DFF;
    cvt_f32_bf16<<<(n + 255) / 256, 256, 0, stream>>>(pr_w, Wpr, n);
  }

  // 2) h = LN1(x)  (bf16)
  ln_bf16<<<NTOK / 8, 256, 0, stream>>>(x, ln1g, ln1b, hbuf, NTOK);

  // 3) qkv = h @ Wqkv^T + b  (bf16 out)
  gemm_bf16<1><<<dim3(3 * DMODEL / 256, NTOK / 64), 256, 0, stream>>>(
      hbuf, Wqkv, in_b, nullptr, qkvB, NTOK, 3 * DMODEL, DMODEL);

  // 4) ctx = flash_attention(q, k, v)
  flash_attn<<<dim3(SEQ / 64, NHEAD, BATCH), 128, 0, stream>>>(qkvB, ctxB);

  // 5) x1 = ctx @ Wout^T + b + x  (f32 out, fused residual)
  gemm_bf16<3><<<dim3(DMODEL / 256, NTOK / 64), 256, 0, stream>>>(
      ctxB, Wout, out_b, x, x1, NTOK, DMODEL, DMODEL);

  // 6) h2 = LN2(x1)  (bf16, reuse hbuf)
  ln_bf16<<<NTOK / 8, 256, 0, stream>>>(x1, ln2g, ln2b, hbuf, NTOK);

  // 7) fcA = quickGELU(h2 @ Wfc^T + b)  (bf16 out)
  gemm_bf16<2><<<dim3(DFF / 256, NTOK / 64), 256, 0, stream>>>(
      hbuf, Wfc, fc_b, nullptr, fcA, NTOK, DFF, DMODEL);

  // 8) out = fcA @ Wpr^T + b + x1  (f32 out, fused residual)
  gemm_bf16<3><<<dim3(DMODEL / 256, NTOK / 64), 256, 0, stream>>>(
      fcA, Wpr, pr_b, x1, (float*)d_out, NTOK, DMODEL, DFF);
}
